// FusedAttention_48808008352213
// MI455X (gfx1250) — matrix operand order repaired and checked by reading, not compiled, not run on hardware
//
#include <hip/hip_runtime.h>
#include <stdint.h>

// ---------------------------------------------------------------------------
// Fused multi-head attention forward, MI455X (gfx1250, wave32, WMMA).
//   ~69 GFLOP vs ~50 MB I/O @ 23.3 TB/s  =>  compute bound by ~2 orders.
//   bf16 WMMA (v_wmma_f32_16x16x32_bf16) for all matmuls, f32 accumulation.
//   Tile staging uses CDNA5 async LDS-direct loads (ASYNCcnt) with ping-pong
//   double buffering: one barrier per K-step, global fetch overlapped with WMMA.
// ---------------------------------------------------------------------------

#define EMBED     1024
#define NHEADS    16
#define HDIM      64
#define BATCHN    2
#define SEQLEN    2048
#define MROWS     (BATCHN * SEQLEN)   // 4096
#define ATT_SCALE 0.125f              // 64^-0.5

typedef __attribute__((ext_vector_type(16))) __bf16 v16bf;
typedef __attribute__((ext_vector_type(8)))  float  v8f;

struct Frag32B { uint4 lo, hi; };     // 32 bytes == one v16bf fragment

static __device__ __forceinline__ unsigned short f2bf_bits(float f) {
  union { float f; uint32_t u; } v; v.f = f;
  uint32_t r = v.u + 0x7FFFu + ((v.u >> 16) & 1u);   // round-to-nearest-even
  return (unsigned short)(r >> 16);
}
static __device__ __forceinline__ __bf16 f2bf(float f) {
  union { unsigned short s; __bf16 h; } o; o.s = f2bf_bits(f);
  return o.h;
}

// Async global->LDS 16B copy (per lane), tracked by ASYNCcnt. LDS operand is
// the wave-relative LDS byte offset == low 32 bits of the generic pointer.
static __device__ __forceinline__ void async_b128(const void* g, void* lds) {
  unsigned l = (unsigned)(uintptr_t)lds;
  asm volatile("global_load_async_to_lds_b128 %0, %1, off"
               :: "v"(l), "v"(g) : "memory");
}
static __device__ __forceinline__ void wait_async() {
  asm volatile("s_wait_asynccnt 0x0" ::: "memory");
}

// Load one 16x32-bf16 A-fragment or 32x16-bf16 B-fragment as two 16B chunks.
static __device__ __forceinline__ v16bf ld_frag(const __bf16* p0, const __bf16* p1) {
  Frag32B r;
  r.lo = *(const uint4*)p0;
  r.hi = *(const uint4*)p1;
  return __builtin_bit_cast(v16bf, r);
}

static __device__ __forceinline__ v8f wmma_bf16(v16bf a, v16bf b, v8f c) {
  // 8-arg form: (neg_a, A, neg_b, B, c_mod, C, reuse_a, reuse_b)
  return __builtin_amdgcn_wmma_f32_16x16x32_bf16(false, a, false, b,
                                                 (short)0, c, false, false);
}

static __device__ __forceinline__ v8f v8f_zero() {
  v8f z = {0.f, 0.f, 0.f, 0.f, 0.f, 0.f, 0.f, 0.f};
  return z;
}

// ---------------------------------------------------------------------------
// Kernel 0: f32 -> bf16 convert (grid-stride)
// ---------------------------------------------------------------------------
__global__ __launch_bounds__(256)
void cvt_f32_bf16(const float* __restrict__ src, unsigned short* __restrict__ dst, int n) {
  int i = blockIdx.x * blockDim.x + threadIdx.x;
  int stride = gridDim.x * blockDim.x;
  for (; i < n; i += stride) dst[i] = f2bf_bits(src[i]);
}

// ---------------------------------------------------------------------------
// Kernel 1: QKV GEMM. A = x_bf16 [4096,1024], W = wqkv_bf16 [3072,1024]
// out[m][n] = sum_k A[m][k] * W[n][k]  (+bias), scattered to q/k/v [B,H,S,D].
// 128x128 workgroup tile, K-step 32, double-buffered async staging.
// ---------------------------------------------------------------------------
__global__ __launch_bounds__(256)
void gemm_qkv(const unsigned short* __restrict__ Au,
              const unsigned short* __restrict__ Wu,
              const float* __restrict__ bias,
              unsigned short* __restrict__ qout,
              unsigned short* __restrict__ kout,
              unsigned short* __restrict__ vout) {
  const __bf16* A = (const __bf16*)Au;
  const __bf16* W = (const __bf16*)Wu;

  __shared__ __align__(16) __bf16 As[2][128][40];   // 40 = 32 + 8 pad
  __shared__ __align__(16) __bf16 Ws[2][128][40];

  const int tid  = threadIdx.x;
  const int wid  = tid >> 5;
  const int lane = tid & 31;
  const int m0 = blockIdx.y * 128;
  const int n0 = blockIdx.x * 128;
  const int waveM = (wid >> 1) * 32;
  const int waveN = (wid & 1) * 64;

  const int r     = lane & 15;        // A row / B col within 16-tile
  const int half  = lane >> 4;
  const int abase = half * 8;         // A fragment K sub-offset
  const int kbB   = half * 16;        // unused; B fragments also use abase

  const int ldRow = tid >> 1;         // 0..127
  const int ldSeg = (tid & 1) * 16;   // 0 or 16 (bf16 elems)

  const __bf16* gA = A + (size_t)(m0 + ldRow) * EMBED + ldSeg;
  const __bf16* gW = W + (size_t)(n0 + ldRow) * EMBED + ldSeg;

  v8f acc[2][4];
  for (int i = 0; i < 2; i++)
    for (int j = 0; j < 4; j++) acc[i][j] = v8f_zero();

  // prologue: stage K-tile 0 into buffer 0
  async_b128(gA,     &As[0][ldRow][ldSeg]);
  async_b128(gA + 8, &As[0][ldRow][ldSeg + 8]);
  async_b128(gW,     &Ws[0][ldRow][ldSeg]);
  async_b128(gW + 8, &Ws[0][ldRow][ldSeg + 8]);

  const int NIT = EMBED / 32;
  for (int it = 0; it < NIT; it++) {
    const int cur = it & 1;
    wait_async();        // current buffer's async loads done (this wave)
    __syncthreads();     // all waves done staging cur & done reading nxt

    if (it + 1 < NIT) {  // overlap next tile fetch with WMMA on cur
      const int nxt = 1 - cur;
      const int k1 = (it + 1) * 32;
      async_b128(gA + k1,     &As[nxt][ldRow][ldSeg]);
      async_b128(gA + k1 + 8, &As[nxt][ldRow][ldSeg + 8]);
      async_b128(gW + k1,     &Ws[nxt][ldRow][ldSeg]);
      async_b128(gW + k1 + 8, &Ws[nxt][ldRow][ldSeg + 8]);
    }

    v16bf aF[2], bF[4];
    for (int i = 0; i < 2; i++)
      aF[i] = ld_frag(&As[cur][waveM + 16 * i + r][abase],
                      &As[cur][waveM + 16 * i + r][16 + abase]);
    for (int j = 0; j < 4; j++)
      bF[j] = ld_frag(&Ws[cur][waveN + 16 * j + r][abase],
                      &Ws[cur][waveN + 16 * j + r][16 + abase]);

    for (int i = 0; i < 2; i++)
      for (int j = 0; j < 4; j++)
        acc[i][j] = wmma_bf16(aF[i], bF[j], acc[i][j]);
  }

  // Epilogue: scatter to q/k/v in [B,H,S,D] bf16. C/D layout: lane holds
  // col = lane&15, rows rowbase..rowbase+7 (rowbase = 8*(lane>>4)).
  const int rowbase = half * 8;
  for (int i = 0; i < 2; i++)
    for (int j = 0; j < 4; j++) {
      const int n = n0 + waveN + 16 * j + r;
      const float bn = bias[n];
      const int which = n >> 10;        // 0=q 1=k 2=v
      const int e = n & 1023;
      const int h = e >> 6;
      const int d = e & 63;
      unsigned short* dst = (which == 0) ? qout : (which == 1) ? kout : vout;
      for (int v = 0; v < 8; v++) {
        const int m = m0 + waveM + 16 * i + rowbase + v;
        const int b = m >> 11;          // / SEQLEN
        const int s = m & 2047;
        dst[(((size_t)(b * NHEADS + h) * SEQLEN) + s) * HDIM + d] =
            f2bf_bits(acc[i][j][v] + bn);
      }
    }
}

// ---------------------------------------------------------------------------
// Kernel 2: flash attention. One workgroup per (b, h, 128-q-row block).
// Each wave owns 16 q rows; Q fragments in registers. Per 64-key block:
// K tile async-staged, V tile transposed through VGPRs; S = Q K^T (WMMA);
// online softmax with half-wave shuffles; P relaid via per-wave LDS slab;
// O += P V (WMMA). K/V tiles ping-pong double buffered.
// ---------------------------------------------------------------------------
__global__ __launch_bounds__(256)
void attn_fwd(const unsigned short* __restrict__ Qu,
              const unsigned short* __restrict__ Ku,
              const unsigned short* __restrict__ Vu,
              unsigned short* __restrict__ ctx) {
  const __bf16* Q = (const __bf16*)Qu;
  const __bf16* K = (const __bf16*)Ku;
  const __bf16* V = (const __bf16*)Vu;

  __shared__ __align__(16) __bf16 Ks[2][64][72];    // [key][d], pad to 72
  __shared__ __align__(16) __bf16 Vt[2][64][72];    // [d][key] (transposed)
  __shared__ __align__(16) __bf16 Ps[8][16][72];    // per-wave P slab

  const int tid  = threadIdx.x;
  const int wid  = tid >> 5;
  const int lane = tid & 31;
  const int b  = blockIdx.z;
  const int h  = blockIdx.y;
  const int q0 = blockIdx.x * 128;

  const size_t headOff = (size_t)(b * NHEADS + h) * SEQLEN * HDIM;
  const __bf16* Qb = Q + headOff;
  const __bf16* Kb = K + headOff;
  const __bf16* Vb = V + headOff;

  const int r     = lane & 15;
  const int half  = lane >> 4;
  const int abase = half * 8;
  const int kbB   = half * 16;
  const int rowbase = half * 8;

  // Q fragments: rows q0 + wid*16 + r, D = 64 -> two 16x32 A fragments
  v16bf qF[2];
  for (int f = 0; f < 2; f++) {
    const __bf16* qr = Qb + (size_t)(q0 + wid * 16 + r) * HDIM + f * 32;
    qF[f] = ld_frag(qr + abase, qr + 16 + abase);
  }

  v8f O[4];
  for (int n = 0; n < 4; n++) O[n] = v8f_zero();
  float mrow[8], lrow[8];
  for (int v = 0; v < 8; v++) { mrow[v] = -3.0e38f; lrow[v] = 0.f; }

  const int ldKey = tid >> 2;          // 0..63
  const int ldSeg = (tid & 3) * 16;    // 0,16,32,48

  const __bf16* gK = Kb + (size_t)ldKey * HDIM + ldSeg;
  const __bf16* gV = Vb + (size_t)ldKey * HDIM + ldSeg;

  // prologue: stage key-block 0 into buffer 0
  async_b128(gK,     &Ks[0][ldKey][ldSeg]);
  async_b128(gK + 8, &Ks[0][ldKey][ldSeg + 8]);
  {
    union { uint4 u[2]; __bf16 hv[16]; } vv;
    const uint4* p = (const uint4*)gV;
    vv.u[0] = p[0]; vv.u[1] = p[1];
    for (int i = 0; i < 16; i++) Vt[0][ldSeg + i][ldKey] = vv.hv[i];
  }

  const int NIT = SEQLEN / 64;
  for (int it = 0; it < NIT; it++) {
    const int cur = it & 1;
    wait_async();
    __syncthreads();

    if (it + 1 < NIT) {   // overlap next K/V block fetch with compute
      const int nxt = 1 - cur;
      const size_t koff = (size_t)(it + 1) * 64 * HDIM;
      async_b128(gK + koff,     &Ks[nxt][ldKey][ldSeg]);
      async_b128(gK + koff + 8, &Ks[nxt][ldKey][ldSeg + 8]);
      union { uint4 u[2]; __bf16 hv[16]; } vv;
      const uint4* p = (const uint4*)(gV + koff);
      vv.u[0] = p[0]; vv.u[1] = p[1];
      for (int i = 0; i < 16; i++) Vt[nxt][ldSeg + i][ldKey] = vv.hv[i];
    }

    // S = Q K^T * scale, 4 key sub-tiles of 16
    v8f s[4];
    for (int j = 0; j < 4; j++) {
      v16bf bk0 = ld_frag(&Ks[cur][16 * j + r][abase],      &Ks[cur][16 * j + r][16 + abase]);
      v16bf bk1 = ld_frag(&Ks[cur][16 * j + r][32 + abase], &Ks[cur][16 * j + r][32 + 16 + abase]);
      v8f cz = v8f_zero();
      cz = wmma_bf16(qF[0], bk0, cz);
      cz = wmma_bf16(qF[1], bk1, cz);
      s[j] = cz * ATT_SCALE;
    }

    // online softmax; row values live across 16 lanes of a half-wave
    float mnew[8];
    for (int v = 0; v < 8; v++) {
      float mx = fmaxf(fmaxf(s[0][v], s[1][v]), fmaxf(s[2][v], s[3][v]));
      for (int off = 1; off < 16; off <<= 1)
        mx = fmaxf(mx, __shfl_xor(mx, off, 16));
      mnew[v] = fmaxf(mrow[v], mx);
    }
    float rsum[8];
    for (int v = 0; v < 8; v++) rsum[v] = 0.f;
    for (int j = 0; j < 4; j++)
      for (int v = 0; v < 8; v++) {
        float p = __expf(s[j][v] - mnew[v]);
        s[j][v] = p;
        rsum[v] += p;
      }
    for (int v = 0; v < 8; v++) {
      for (int off = 1; off < 16; off <<= 1)
        rsum[v] += __shfl_xor(rsum[v], off, 16);
      float corr = __expf(mrow[v] - mnew[v]);
      lrow[v] = lrow[v] * corr + rsum[v];
      mrow[v] = mnew[v];
      for (int n = 0; n < 4; n++) O[n][v] *= corr;
    }

    // P: C/D layout -> A layout via wave-private LDS slab (in-order DS)
    for (int j = 0; j < 4; j++)
      for (int v = 0; v < 8; v++)
        Ps[wid][rowbase + v][16 * j + r] = f2bf(s[j][v]);

    v16bf pF[2];
    for (int f = 0; f < 2; f++)
      pF[f] = ld_frag(&Ps[wid][r][f * 32 + abase],
                      &Ps[wid][r][f * 32 + 16 + abase]);

    // O += P * V   (B fragments from transposed V tile: contiguous reads)
    for (int n = 0; n < 4; n++) {
      v16bf bv0 = ld_frag(&Vt[cur][n * 16 + r][abase],      &Vt[cur][n * 16 + r][16 + abase]);
      v16bf bv1 = ld_frag(&Vt[cur][n * 16 + r][32 + abase], &Vt[cur][n * 16 + r][32 + 16 + abase]);
      O[n] = wmma_bf16(pF[0], bv0, O[n]);
      O[n] = wmma_bf16(pF[1], bv1, O[n]);
    }
  }

  // normalize and write ctx in [B,S,H,D] (= [B,S,E]) bf16
  for (int v = 0; v < 8; v++) {
    const float inv = 1.0f / lrow[v];
    const int srow = q0 + wid * 16 + rowbase + v;
    unsigned short* dstc =
        ctx + ((size_t)(b * SEQLEN + srow) * NHEADS + h) * HDIM;
    for (int n = 0; n < 4; n++)
      dstc[n * 16 + r] = f2bf_bits(O[n][v] * inv);
  }
}

// ---------------------------------------------------------------------------
// Kernel 3: output projection. A = ctx_bf16 [4096,1024], W = wout_bf16 [1024,1024]
// out f32 [4096,1024] = A W^T + bias.  Same tiling as gemm_qkv.
// ---------------------------------------------------------------------------
__global__ __launch_bounds__(256)
void gemm_out(const unsigned short* __restrict__ Au,
              const unsigned short* __restrict__ Wu,
              const float* __restrict__ bias,
              float* __restrict__ outp) {
  const __bf16* A = (const __bf16*)Au;
  const __bf16* W = (const __bf16*)Wu;

  __shared__ __align__(16) __bf16 As[2][128][40];
  __shared__ __align__(16) __bf16 Ws[2][128][40];

  const int tid  = threadIdx.x;
  const int wid  = tid >> 5;
  const int lane = tid & 31;
  const int m0 = blockIdx.y * 128;
  const int n0 = blockIdx.x * 128;
  const int waveM = (wid >> 1) * 32;
  const int waveN = (wid & 1) * 64;

  const int r     = lane & 15;
  const int half  = lane >> 4;
  const int abase = half * 8;
  const int kbB   = half * 16;

  const int ldRow = tid >> 1;
  const int ldSeg = (tid & 1) * 16;

  const __bf16* gA = A + (size_t)(m0 + ldRow) * EMBED + ldSeg;
  const __bf16* gW = W + (size_t)(n0 + ldRow) * EMBED + ldSeg;

  v8f acc[2][4];
  for (int i = 0; i < 2; i++)
    for (int j = 0; j < 4; j++) acc[i][j] = v8f_zero();

  async_b128(gA,     &As[0][ldRow][ldSeg]);
  async_b128(gA + 8, &As[0][ldRow][ldSeg + 8]);
  async_b128(gW,     &Ws[0][ldRow][ldSeg]);
  async_b128(gW + 8, &Ws[0][ldRow][ldSeg + 8]);

  const int NIT = EMBED / 32;
  for (int it = 0; it < NIT; it++) {
    const int cur = it & 1;
    wait_async();
    __syncthreads();

    if (it + 1 < NIT) {
      const int nxt = 1 - cur;
      const int k1 = (it + 1) * 32;
      async_b128(gA + k1,     &As[nxt][ldRow][ldSeg]);
      async_b128(gA + k1 + 8, &As[nxt][ldRow][ldSeg + 8]);
      async_b128(gW + k1,     &Ws[nxt][ldRow][ldSeg]);
      async_b128(gW + k1 + 8, &Ws[nxt][ldRow][ldSeg + 8]);
    }

    v16bf aF[2], bF[4];
    for (int i = 0; i < 2; i++)
      aF[i] = ld_frag(&As[cur][waveM + 16 * i + r][abase],
                      &As[cur][waveM + 16 * i + r][16 + abase]);
    for (int j = 0; j < 4; j++)
      bF[j] = ld_frag(&Ws[cur][waveN + 16 * j + r][abase],
                      &Ws[cur][waveN + 16 * j + r][16 + abase]);

    for (int i = 0; i < 2; i++)
      for (int j = 0; j < 4; j++)
        acc[i][j] = wmma_bf16(aF[i], bF[j], acc[i][j]);
  }

  const int rowbase = half * 8;
  for (int i = 0; i < 2; i++)
    for (int j = 0; j < 4; j++) {
      const int n = n0 + waveN + 16 * j + r;
      const float bn = bias[n];
      for (int v = 0; v < 8; v++) {
        const int m = m0 + waveM + 16 * i + rowbase + v;
        outp[(size_t)m * EMBED + n] = acc[i][j][v] + bn;
      }
    }
}

// ---------------------------------------------------------------------------
// Launch
// ---------------------------------------------------------------------------
extern "C" void kernel_launch(void* const* d_in, const int* in_sizes, int n_in,
                              void* d_out, int out_size, void* d_ws, size_t ws_size,
                              hipStream_t stream) {
  const float* x        = (const float*)d_in[0];
  const float* w_qkv    = (const float*)d_in[1];
  const float* w_out    = (const float*)d_in[2];
  const float* bias_qkv = (const float*)d_in[3];
  const float* bias_out = (const float*)d_in[4];
  float* out = (float*)d_out;

  // workspace layout (bf16 buffers), ~48 MB total
  char* ws = (char*)d_ws;
  size_t off = 0;
  auto alloc = [&](size_t bytes) -> void* {
    void* p = ws + off;
    off += (bytes + 255) & ~(size_t)255;
    return p;
  };
  unsigned short* x_bf    = (unsigned short*)alloc((size_t)MROWS * EMBED * 2);
  unsigned short* wqkv_bf = (unsigned short*)alloc((size_t)3 * EMBED * EMBED * 2);
  unsigned short* wout_bf = (unsigned short*)alloc((size_t)EMBED * EMBED * 2);
  unsigned short* q_ws    = (unsigned short*)alloc((size_t)MROWS * EMBED * 2);
  unsigned short* k_ws    = (unsigned short*)alloc((size_t)MROWS * EMBED * 2);
  unsigned short* v_ws    = (unsigned short*)alloc((size_t)MROWS * EMBED * 2);
  unsigned short* ctx_ws  = (unsigned short*)alloc((size_t)MROWS * EMBED * 2);

  cvt_f32_bf16<<<512, 256, 0, stream>>>(x, x_bf, MROWS * EMBED);
  cvt_f32_bf16<<<512, 256, 0, stream>>>(w_qkv, wqkv_bf, 3 * EMBED * EMBED);
  cvt_f32_bf16<<<256, 256, 0, stream>>>(w_out, wout_bf, EMBED * EMBED);

  gemm_qkv<<<dim3(3 * EMBED / 128, MROWS / 128), 256, 0, stream>>>(
      x_bf, wqkv_bf, bias_qkv, q_ws, k_ws, v_ws);

  attn_fwd<<<dim3(SEQLEN / 128, NHEADS, BATCHN), 256, 0, stream>>>(
      q_ws, k_ws, v_ws, ctx_ws);

  gemm_out<<<dim3(EMBED / 128, MROWS / 128), 256, 0, stream>>>(
      ctx_ws, wout_bf, bias_out, out);
}
